// PAM_86105504350481
// MI455X (gfx1250) — matrix operand order repaired and checked by reading, not compiled, not run on hardware
//
#include <hip/hip_runtime.h>
#include <hip/hip_bf16.h>

// PAM attention (DANet-style): projection GEMM (WMMA bf16) + flash-attention
// (WMMA bf16, online softmax) on gfx1250 wave32.
// CDNA5 paths: v_wmma_f32_16x16x32_bf16 core math, global_load_async_to_lds_b128
// (ASYNCcnt) staging of Q/K/V tiles, ds_load_b128 fragment feeds.
// Workspace: 40 MB of d_ws for bf16 Q (B,N,64), K^T (B,N,64), V^T (B,C,N).

typedef unsigned short u16;
typedef __attribute__((ext_vector_type(16))) __bf16 v16bf;
typedef __attribute__((ext_vector_type(8)))  float  v8f;

#define BATCH 8
#define CCH   512
#define NPOS  4096
#define MIDC  64

__device__ __forceinline__ u16 f2bf(float f) {
    union { float f; unsigned u; } x; x.f = f;
    unsigned r = x.u + 0x7FFFu + ((x.u >> 16) & 1u);   // round-to-nearest-even
    return (u16)(r >> 16);
}

__device__ __forceinline__ v8f v8f_zero() {
    v8f z = {0.f,0.f,0.f,0.f,0.f,0.f,0.f,0.f};
    return z;
}

// Async copy of 16 bytes global -> LDS, tracked by ASYNCcnt (no VGPR data).
// GVS mode: 64-bit SGPR base + 32-bit per-lane byte offset.
__device__ __forceinline__ void async_b128(unsigned lds_byte, const void* sbase,
                                           unsigned gbl_byte) {
    asm volatile("global_load_async_to_lds_b128 %0, %1, %2 offset:0"
                 :
                 : "v"(lds_byte), "v"(gbl_byte), "s"(sbase)
                 : "memory");
}
__device__ __forceinline__ void wait_async0() {
    asm volatile("s_wait_asynccnt 0" ::: "memory");
}

// A-matrix fragment (16x32 bf16, M x K). rowbase points at LDS row (K-chunk
// start, K contiguous). Lanes 0-15: M=lane, K = {half*8..+7, 16+half*8..+7}.
__device__ __forceinline__ v16bf load_frag_a(const u16* rowbase, int half) {
    union { uint4 q[2]; v16bf v; } r;
    r.q[0] = *(const uint4*)(rowbase + half * 8);
    r.q[1] = *(const uint4*)(rowbase + 16 + half * 8);
    return r.v;
}

// B-matrix fragment (32x16 bf16, K x N). rowbase points at LDS row for this
// lane's column (K contiguous). Lanes 0-15: N=lane, K = {half*8..+7, 16+half*8..+7}.
__device__ __forceinline__ v16bf load_frag_b(const u16* rowbase, int half) {
    union { uint4 q[2]; v16bf v; } r;
    r.q[0] = *(const uint4*)(rowbase + half * 8);
    r.q[1] = *(const uint4*)(rowbase + 16 + half * 8);
    return r.v;
}

__device__ __forceinline__ v8f wmma_bf16(v16bf a, v16bf b, v8f c) {
    return __builtin_amdgcn_wmma_f32_16x16x32_bf16(
        /*neg_a=*/false, a, /*neg_b=*/false, b,
        /*c_mod=*/(short)0, c, /*reuse_a=*/false, /*reuse_b=*/false);
}

// ---------------------------------------------------------------------------
// Kernel 1: projections.  Per block: one batch, 32 n-rows; output 640 columns
// (Q mid=64 | K mid=64 | V C=512) = (32x512)@(512x640) GEMM in K=32 chunks.
// 8 waves x 10 tiles of 16x16, f32 accum.  Q/K stored (N,64); V stored
// transposed (C,N) so the attention kernel can async-copy it untransposed.
// ---------------------------------------------------------------------------
__global__ __launch_bounds__(256) void pam_proj_kernel(
    const float* __restrict__ x, const float* __restrict__ Wb,
    const float* __restrict__ Wc, const float* __restrict__ Wd,
    u16* __restrict__ bq, u16* __restrict__ kk, u16* __restrict__ vvt) {
    extern __shared__ __align__(16) char smem[];
    u16* As = (u16*)smem;          // [32 n][32 c]
    u16* Ws = As + 32 * 32;        // [640 o][32 c]

    const int tid  = threadIdx.x;
    const int wave = tid >> 5;
    const int lane = tid & 31;
    const int lcol = lane & 15;
    const int half = lane >> 4;
    const int b     = blockIdx.x >> 7;          // 8 batches
    const int nbase = (blockIdx.x & 127) * 32;  // 128 n-tiles/batch

    v8f acc[10];
#pragma unroll
    for (int t = 0; t < 10; ++t) acc[t] = v8f_zero();

    const size_t xb = (size_t)b * CCH * NPOS;

    for (int kc = 0; kc < CCH / 32; ++kc) {
        const int cbase = kc * 32;
        __syncthreads();
        // stage x^T tile: As[n][c]  (global reads coalesced over n)
        for (int i = tid; i < 32 * 32; i += 256) {
            int nn = i & 31, cc = i >> 5;
            As[nn * 32 + cc] =
                f2bf(x[xb + (size_t)(cbase + cc) * NPOS + nbase + nn]);
        }
        // stage concatenated weights: Ws[o][c]
        for (int i = tid; i < 640 * 32; i += 256) {
            int cc = i & 31, o = i >> 5;
            float wv;
            if (o < 64)        wv = Wb[o * CCH + cbase + cc];
            else if (o < 128)  wv = Wc[(o - 64) * CCH + cbase + cc];
            else               wv = Wd[(o - 128) * CCH + cbase + cc];
            Ws[o * 32 + cc] = f2bf(wv);
        }
        __syncthreads();
#pragma unroll
        for (int t = 0; t < 10; ++t) {
            int tile = wave * 10 + t;
            int r = tile / 40, ci = tile % 40;
            v16bf a = load_frag_a(&As[(r * 16 + lcol) * 32], half);
            v16bf w = load_frag_b(&Ws[(ci * 16 + lcol) * 32], half);
            acc[t] = wmma_bf16(a, w, acc[t]);
        }
    }

#pragma unroll
    for (int t = 0; t < 10; ++t) {
        int tile = wave * 10 + t;
        int r = tile / 40, ci = tile % 40;
        int gco = ci * 16 + lcol;
        if (gco >= 128) {
            // V: lane owns 8 consecutive n at fixed channel o -> packed store
            int o  = gco - 128;
            int n0 = nbase + r * 16 + half * 8;
            union { uint4 q; u16 h[8]; } pk;
#pragma unroll
            for (int j = 0; j < 8; ++j) pk.h[j] = f2bf(acc[t][j]);
            *(uint4*)&vvt[((size_t)b * CCH + o) * NPOS + n0] = pk.q;
        } else {
#pragma unroll
            for (int j = 0; j < 8; ++j) {
                int n = nbase + r * 16 + j + half * 8;
                u16 hv = f2bf(acc[t][j]);
                size_t rowi = (size_t)b * NPOS + n;
                if (gco < 64) bq[rowi * MIDC + gco] = hv;
                else          kk[rowi * MIDC + (gco - 64)] = hv;
            }
        }
    }
}

// ---------------------------------------------------------------------------
// Kernel 2: flash attention.  Per block: one batch, 64 query rows.  Loop over
// 64-key tiles: async-stage K,V tiles into LDS, S = Q K^T (WMMA), online
// softmax (m,l in LDS), rescale 64x512 f32 accumulators, O += P V (WMMA).
// Final: O/l * gamma + x, stored into (B, C, N) with float4 ops.
// ---------------------------------------------------------------------------
__global__ __launch_bounds__(256) void pam_attn_kernel(
    const float* __restrict__ x, const u16* __restrict__ bq,
    const u16* __restrict__ kk, const u16* __restrict__ vvt,
    const float* __restrict__ gamma, float* __restrict__ out) {
    extern __shared__ __align__(16) char smem[];
    u16*   Qs   = (u16*)smem;              // [64 n][64 m]
    u16*   Ks   = Qs + 64 * 64;            // [64 k][64 m]
    u16*   Pb   = Ks + 64 * 64;            // [64 n][64 k]  exp(S - m) in bf16
    u16*   Vt   = Pb + 64 * 64;            // [512 o][64 k]
    float* Sf   = (float*)(Vt + 512 * 64); // [64 n][64 k] raw scores
    float* mrun = Sf + 64 * 64;            // running max  [64]
    float* lrun = mrun + 64;               // running sum  [64]
    float* rsc  = lrun + 64;               // rescale      [64]

    const unsigned lds0    = (unsigned)(size_t)smem;  // LDS byte base
    const unsigned QS_OFF  = lds0;
    const unsigned KS_OFF  = lds0 + 64 * 64 * 2;
    const unsigned VT_OFF  = lds0 + 3 * 64 * 64 * 2;

    const int tid  = threadIdx.x;
    const int wave = tid >> 5;
    const int lane = tid & 31;
    const int lcol = lane & 15;
    const int half = lane >> 4;
    const int b     = blockIdx.x >> 6;         // 8 batches
    const int nbase = (blockIdx.x & 63) * 64;  // 64 q-tiles/batch

    // async-stage Q tile (contiguous 8 KB): 512 b128 units, 2 per thread
    const u16* qsrc = bq + ((size_t)b * NPOS + nbase) * MIDC;
#pragma unroll
    for (int u = 0; u < 2; ++u) {
        unsigned unit = tid + u * 256;
        async_b128(QS_OFF + unit * 16, qsrc, unit * 16);
    }
    if (tid < 64) { mrun[tid] = -1e30f; lrun[tid] = 0.f; }

    v8f acc[16];
#pragma unroll
    for (int t = 0; t < 16; ++t) acc[t] = v8f_zero();

    const int rt      = wave >> 1;         // query row-tile of this wave
    const int colbase = (wave & 1) * 256;  // value-column half of this wave

    const u16* vbase = vvt + (size_t)b * CCH * NPOS;   // (C, N) for this batch

    for (int kt = 0; kt < NPOS / 64; ++kt) {
        const int kbase = kt * 64;
        __syncthreads();  // previous P@V done before restaging
        // async-stage K tile (contiguous 8 KB)
        const u16* ksrc = kk + ((size_t)b * NPOS + kbase) * MIDC;
#pragma unroll
        for (int u = 0; u < 2; ++u) {
            unsigned unit = tid + u * 256;
            async_b128(KS_OFF + unit * 16, ksrc, unit * 16);
        }
        // async-stage V tile: Vt[o][k] <- vvt[o][kbase+k], 64 KB, 16/thread
        const u16* vsrc = vbase + kbase;   // row o at byte offset o*NPOS*2
#pragma unroll
        for (int u = 0; u < 16; ++u) {
            unsigned unit = tid + u * 256;         // 0..4095
            unsigned o    = unit >> 3;
            unsigned kseg = unit & 7;
            async_b128(VT_OFF + unit * 16, vsrc, o * (NPOS * 2) + kseg * 16);
        }
        wait_async0();
        __syncthreads();

        // S = Q K^T : 16 score tiles, 2 per wave, K-dim = mid = 64
#pragma unroll
        for (int s = 0; s < 2; ++s) {
            int tileid = wave * 2 + s;
            int srt = tileid >> 2, sct = tileid & 3;
            v8f sv = v8f_zero();
#pragma unroll
            for (int kc = 0; kc < 2; ++kc) {
                v16bf a  = load_frag_a(&Qs[(srt * 16 + lcol) * 64 + kc * 32], half);
                v16bf bb = load_frag_b(&Ks[(sct * 16 + lcol) * 64 + kc * 32], half);
                sv = wmma_bf16(a, bb, sv);
            }
            int col = sct * 16 + lcol;
#pragma unroll
            for (int j = 0; j < 8; ++j)
                Sf[(srt * 16 + j + half * 8) * 64 + col] = sv[j];
        }
        __syncthreads();

        // online softmax, one thread per query row
        if (tid < 64) {
            int r = tid;
            float mold = mrun[r];
            float tmax = -1e30f;
            for (int k2 = 0; k2 < 64; ++k2) tmax = fmaxf(tmax, Sf[r * 64 + k2]);
            float mnew = fmaxf(mold, tmax);
            float sc   = __expf(mold - mnew);
            float ssum = 0.f;
            for (int k2 = 0; k2 < 64; ++k2) {
                float p = __expf(Sf[r * 64 + k2] - mnew);
                ssum += p;
                Pb[r * 64 + k2] = f2bf(p);
            }
            lrun[r] = lrun[r] * sc + ssum;
            mrun[r] = mnew;
            rsc[r]  = sc;
        }
        __syncthreads();

        // rescale accumulators by exp(m_old - m_new)
#pragma unroll
        for (int j = 0; j < 8; ++j) {
            float sc = rsc[rt * 16 + j + half * 8];
#pragma unroll
            for (int t = 0; t < 16; ++t) acc[t][j] *= sc;
        }
        // O += P V : per wave 16 col-tiles, K-dim = 64 keys
        v16bf a0 = load_frag_a(&Pb[(rt * 16 + lcol) * 64 + 0], half);
        v16bf a1 = load_frag_a(&Pb[(rt * 16 + lcol) * 64 + 32], half);
#pragma unroll
        for (int t = 0; t < 16; ++t) {
            int o = colbase + t * 16 + lcol;
            v16bf b0 = load_frag_b(&Vt[o * 64 + 0], half);
            v16bf b1 = load_frag_b(&Vt[o * 64 + 32], half);
            acc[t] = wmma_bf16(a0, b0, acc[t]);
            acc[t] = wmma_bf16(a1, b1, acc[t]);
        }
    }
    __syncthreads();

    // epilogue: lane owns 8 consecutive n at fixed channel o -> float4 ops
    const float g  = gamma[0];
    const int  rl0 = rt * 16 + half * 8;   // first of 8 consecutive local rows
#pragma unroll
    for (int t = 0; t < 16; ++t) {
        int o = colbase + t * 16 + lcol;
        size_t base = ((size_t)b * CCH + o) * NPOS + nbase + rl0;
        float4 xi0 = *(const float4*)&x[base];
        float4 xi1 = *(const float4*)&x[base + 4];
        float4 r0, r1;
        r0.x = g * (acc[t][0] / lrun[rl0 + 0]) + xi0.x;
        r0.y = g * (acc[t][1] / lrun[rl0 + 1]) + xi0.y;
        r0.z = g * (acc[t][2] / lrun[rl0 + 2]) + xi0.z;
        r0.w = g * (acc[t][3] / lrun[rl0 + 3]) + xi0.w;
        r1.x = g * (acc[t][4] / lrun[rl0 + 4]) + xi1.x;
        r1.y = g * (acc[t][5] / lrun[rl0 + 5]) + xi1.y;
        r1.z = g * (acc[t][6] / lrun[rl0 + 6]) + xi1.z;
        r1.w = g * (acc[t][7] / lrun[rl0 + 7]) + xi1.w;
        *(float4*)&out[base]     = r0;
        *(float4*)&out[base + 4] = r1;
    }
}

extern "C" void kernel_launch(void* const* d_in, const int* in_sizes, int n_in,
                              void* d_out, int out_size, void* d_ws, size_t ws_size,
                              hipStream_t stream) {
    (void)in_sizes; (void)n_in; (void)out_size; (void)ws_size;
    const float* x     = (const float*)d_in[0];
    const float* Wb    = (const float*)d_in[1];
    const float* Wc    = (const float*)d_in[2];
    const float* Wd    = (const float*)d_in[3];
    const float* gamma = (const float*)d_in[4];
    float* out = (float*)d_out;

    // workspace: bf16 Q (B,N,64) | K^T (B,N,64) | V^T (B,C,N)  = 40 MB
    u16* bq  = (u16*)d_ws;
    u16* kk  = bq + (size_t)BATCH * NPOS * MIDC;
    u16* vvt = kk + (size_t)BATCH * NPOS * MIDC;

    const size_t lds1 = (32 * 32 + 640 * 32) * sizeof(u16);                 // 43 KB
    const size_t lds2 = (3 * 64 * 64 + 512 * 64) * sizeof(u16)
                      + (64 * 64 + 3 * 64) * sizeof(float);                 // 107 KB

    pam_proj_kernel<<<dim3(BATCH * (NPOS / 32)), dim3(256), lds1, stream>>>(
        x, Wb, Wc, Wd, bq, kk, vvt);
    pam_attn_kernel<<<dim3(BATCH * (NPOS / 64)), dim3(256), lds2, stream>>>(
        x, bq, kk, vvt, gamma, out);
}
